// MPNN_3917010174729
// MI455X (gfx1250) — compile-verified, matrix operand order checked by reading, not run on hardware
//
#include <hip/hip_runtime.h>

#define NN 50000
#define NE 800000
// DIN=64 EIN=32 DOUT=64 EOUT=64 H=128 ; e_in = 160, n_in = 128

typedef __attribute__((ext_vector_type(16))) __bf16 bf16x16;
typedef __attribute__((ext_vector_type(8)))  __bf16 bf16x8;
typedef __attribute__((ext_vector_type(8)))  float  f32x8;

// ---- helpers -------------------------------------------------------------

__device__ __forceinline__ unsigned short f2bf(float f) {
    unsigned int u = __float_as_uint(f);
    unsigned int r = 0x7FFFu + ((u >> 16) & 1u);
    return (unsigned short)((u + r) >> 16);
}
__device__ __forceinline__ float bf2f(unsigned short b) {
    return __uint_as_float(((unsigned int)b) << 16);
}
// monotonic unsigned mapping of float for atomicMax-based segment max
__device__ __forceinline__ unsigned fmap(float f) {
    unsigned u = __float_as_uint(f);
    return (u & 0x80000000u) ? ~u : (u | 0x80000000u);
}
__device__ __forceinline__ float funmap(unsigned m) {
    unsigned u = (m & 0x80000000u) ? (m ^ 0x80000000u) : ~m;
    return __uint_as_float(u);
}
#define MAP_NEG_INF 0x007FFFFFu  // fmap(-inf)

// Load a 16-bit A/B WMMA fragment.  Per ISA §7.12.2 (16-bit 16x32 A):
// lanes 0-15 hold K octets {0..7, 16..23}, lanes 16-31 hold {8..15, 24..31};
// row (A) / col (B) index = lane&15.  base = &data[(row_or_col)*stride + kb].
__device__ __forceinline__ bf16x16 load_frag(const unsigned short* base, int half) {
    bf16x16 v;
    ((bf16x8*)&v)[0] = *(const bf16x8*)(base + 8 * half);
    ((bf16x8*)&v)[1] = *(const bf16x8*)(base + 16 + 8 * half);
    return v;
}

__device__ __forceinline__ f32x8 wmma_bf16(bf16x16 a, bf16x16 b, f32x8 c) {
    return __builtin_amdgcn_wmma_f32_16x16x32_bf16(
        /*neg_a=*/false, a, /*neg_b=*/false, b,
        /*c_mod=*/(short)0, c, /*reuse_a=*/false, /*reuse_b=*/false);
}

// ---- kernel 0: weight transpose + f32->bf16 ------------------------------
// W[K][N] f32 row-major  ->  WT[N][K] bf16 row-major (B-fragment friendly)
__global__ void mpnn_wprep(const float* __restrict__ W,
                           unsigned short* __restrict__ WT, int K, int N) {
    int idx = blockIdx.x * 256 + threadIdx.x;
    if (idx >= K * N) return;
    int k = idx / N, n = idx - k * N;
    WT[n * K + k] = f2bf(W[idx]);
}

// ---- kernel 1: init scratch ---------------------------------------------
__global__ void mpnn_init(float* __restrict__ agg, float* __restrict__ denom,
                          unsigned* __restrict__ mmax) {
    int i = blockIdx.x * 256 + threadIdx.x;
    if (i < NN * 64) agg[i] = 0.0f;
    if (i < NN) { denom[i] = 0.0f; mmax[i] = MAP_NEG_INF; }
}

// ---- kernel 2: edge MLPs + attention logit + segment max -----------------
// block = 256 threads (8 waves), 64 edges per block.
__global__ __launch_bounds__(256) void mpnn_edge(
    const float* __restrict__ nf, const float* __restrict__ ef,
    const int* __restrict__ src, const int* __restrict__ dst,
    const unsigned short* __restrict__ wtWe1, const float* __restrict__ be1,
    const unsigned short* __restrict__ wtWa1, const float* __restrict__ ba1,
    const unsigned short* __restrict__ wtWe2, const float* __restrict__ be2,
    const float* __restrict__ Wa2, const float* __restrict__ ba2,
    float* __restrict__ uh_e, float* __restrict__ logits,
    unsigned* __restrict__ mmax) {
    __shared__ unsigned short sA[64 * 160];   // concat input, bf16
    __shared__ unsigned short sH[64 * 128];   // edge hidden (post-ReLU), bf16
    __shared__ unsigned short sHa[64 * 128];  // attn hidden (post-ReLU), bf16
    __shared__ int sSrc[64], sDst[64];

    const int tid = threadIdx.x;
    const int e0 = blockIdx.x * 64;
    if (tid < 64) {
        int e = e0 + tid;
        sSrc[tid] = src[e];
        sDst[tid] = dst[e];
    }
    __syncthreads();

    // gather + concat [nf[src] | nf[dst] | ef] -> bf16 LDS
    for (int idx = tid; idx < 64 * 160; idx += 256) {
        int r = idx / 160, c = idx - r * 160;
        float v;
        if (c < 64)       v = nf[sSrc[r] * 64 + c];
        else if (c < 128) v = nf[sDst[r] * 64 + (c - 64)];
        else              v = ef[(e0 + r) * 32 + (c - 128)];
        sA[idx] = f2bf(v);
    }
    __syncthreads();

    const int wave = tid >> 5, lane = tid & 31;
    const int half = lane >> 4, nl = lane & 15;

    // ---- layer 1: [64,160] x 160->128, waves 0-3: We1 -> sH, 4-7: Wa1 -> sHa
    {
        const int mSub = wave & 3;
        const unsigned short* WT = (wave < 4) ? wtWe1 : wtWa1;  // [128][160]
        const float* bias = (wave < 4) ? be1 : ba1;
        unsigned short* HO = (wave < 4) ? sH : sHa;
        f32x8 acc[8];
#pragma unroll
        for (int n = 0; n < 8; ++n) acc[n] = (f32x8)0.0f;
#pragma unroll
        for (int kb = 0; kb < 160; kb += 32) {
            bf16x16 aF = load_frag(&sA[(mSub * 16 + nl) * 160 + kb], half);
#pragma unroll
            for (int n = 0; n < 8; ++n) {
                bf16x16 bF = load_frag(&WT[(n * 16 + nl) * 160 + kb], half);
                acc[n] = wmma_bf16(aF, bF, acc[n]);
            }
        }
#pragma unroll
        for (int n = 0; n < 8; ++n) {
            int col = n * 16 + nl;
            float bv = bias[col];
#pragma unroll
            for (int v = 0; v < 8; ++v) {
                int row = mSub * 16 + v + 8 * half;  // C layout: M = v + 8*half
                float x = acc[n][v] + bv;
                HO[row * 128 + col] = f2bf(x > 0.0f ? x : 0.0f);
            }
        }
    }
    __syncthreads();

    // ---- layer 2 (edge): [64,128] x 128->64, 16 tiles, 2 per wave
#pragma unroll
    for (int t = wave * 2; t < wave * 2 + 2; ++t) {
        int m2 = (t >> 2) * 16, n2 = (t & 3) * 16;
        f32x8 c2 = (f32x8)0.0f;
#pragma unroll
        for (int kb = 0; kb < 128; kb += 32) {
            bf16x16 aF = load_frag(&sH[(m2 + nl) * 128 + kb], half);
            bf16x16 bF = load_frag(&wtWe2[(n2 + nl) * 128 + kb], half);
            c2 = wmma_bf16(aF, bF, c2);
        }
        int col = n2 + nl;
        float bv = be2[col];
#pragma unroll
        for (int v = 0; v < 8; ++v) {
            int e = e0 + m2 + v + 8 * half;
            uh_e[e * 64 + col] = c2[v] + bv;
        }
    }

    // ---- attention head: logit = dot(sHa_row, Wa2) + ba2 ; segment max
    if (tid < 64) {
        float s = ba2[0];
#pragma unroll 8
        for (int k = 0; k < 128; ++k) s += bf2f(sHa[tid * 128 + k]) * Wa2[k];
        int e = e0 + tid;
        logits[e] = s;
        atomicMax(&mmax[sDst[tid]], fmap(s));
    }
}

// ---- kernel 3: ex = exp(logit - m[dst]); denom[dst] += ex ----------------
__global__ void mpnn_softmax(const float* __restrict__ logits,
                             const int* __restrict__ dst,
                             const unsigned* __restrict__ mmax,
                             float* __restrict__ exv,
                             float* __restrict__ denom) {
    int e = blockIdx.x * 256 + threadIdx.x;
    if (e >= NE) return;
    int d = dst[e];
    float m = funmap(mmax[d]);
    if (!isfinite(m)) m = 0.0f;
    float v = __expf(logits[e] - m);
    exv[e] = v;
    atomicAdd(&denom[d], v);
}

// ---- kernel 4: agg[dst] += uh_e * attn (one thread per edge-channel) -----
__global__ void mpnn_agg(const float* __restrict__ exv,
                         const float* __restrict__ denom,
                         const int* __restrict__ dst,
                         const float* __restrict__ uh_e,
                         float* __restrict__ agg) {
    int gid = blockIdx.x * 256 + threadIdx.x;  // < NE*64
    int e = gid >> 6, c = gid & 63;
    int d = dst[e];
    float attn = exv[e] / fmaxf(denom[d], 1e-38f);
    atomicAdd(&agg[d * 64 + c], uh_e[gid] * attn);
}

// ---- kernel 5: node MLP on concat(agg, nf) -------------------------------
__global__ __launch_bounds__(256) void mpnn_node(
    const float* __restrict__ agg, const float* __restrict__ nf,
    const unsigned short* __restrict__ wtWn1, const float* __restrict__ bn1,
    const unsigned short* __restrict__ wtWn2, const float* __restrict__ bn2,
    float* __restrict__ uh_n) {
    __shared__ unsigned short sA[64 * 128];
    __shared__ unsigned short sH[64 * 128];

    const int tid = threadIdx.x;
    const int n0 = blockIdx.x * 64;

    for (int idx = tid; idx < 64 * 128; idx += 256) {
        int r = idx >> 7, c = idx & 127;
        int n = n0 + r;
        float v = 0.0f;
        if (n < NN) v = (c < 64) ? agg[n * 64 + c] : nf[n * 64 + (c - 64)];
        sA[idx] = f2bf(v);
    }
    __syncthreads();

    const int wave = tid >> 5, lane = tid & 31;
    const int half = lane >> 4, nl = lane & 15;

    // layer 1: [64,128] x 128->128 ; 32 tiles, 4 per wave
    {
        const int mSub = wave & 3;
        const int nG = wave >> 2;
        f32x8 acc[4];
#pragma unroll
        for (int q = 0; q < 4; ++q) acc[q] = (f32x8)0.0f;
#pragma unroll
        for (int kb = 0; kb < 128; kb += 32) {
            bf16x16 aF = load_frag(&sA[(mSub * 16 + nl) * 128 + kb], half);
#pragma unroll
            for (int q = 0; q < 4; ++q) {
                int nSub = nG * 4 + q;
                bf16x16 bF = load_frag(&wtWn1[(nSub * 16 + nl) * 128 + kb], half);
                acc[q] = wmma_bf16(aF, bF, acc[q]);
            }
        }
        __syncthreads();  // done reading sA before anyone could race (none do), cheap
#pragma unroll
        for (int q = 0; q < 4; ++q) {
            int col = (nG * 4 + q) * 16 + nl;
            float bv = bn1[col];
#pragma unroll
            for (int v = 0; v < 8; ++v) {
                int row = mSub * 16 + v + 8 * half;
                float x = acc[q][v] + bv;
                sH[row * 128 + col] = f2bf(x > 0.0f ? x : 0.0f);
            }
        }
    }
    __syncthreads();

    // layer 2: [64,128] x 128->64 ; 16 tiles, 2 per wave
#pragma unroll
    for (int t = wave * 2; t < wave * 2 + 2; ++t) {
        int m2 = (t >> 2) * 16, n2 = (t & 3) * 16;
        f32x8 c2 = (f32x8)0.0f;
#pragma unroll
        for (int kb = 0; kb < 128; kb += 32) {
            bf16x16 aF = load_frag(&sH[(m2 + nl) * 128 + kb], half);
            bf16x16 bF = load_frag(&wtWn2[(n2 + nl) * 128 + kb], half);
            c2 = wmma_bf16(aF, bF, c2);
        }
        int col = n2 + nl;
        float bv = bn2[col];
#pragma unroll
        for (int v = 0; v < 8; ++v) {
            int n = n0 + m2 + v + 8 * half;
            if (n < NN) uh_n[n * 64 + col] = c2[v] + bv;
        }
    }
}

// ---- launch --------------------------------------------------------------
extern "C" void kernel_launch(void* const* d_in, const int* in_sizes, int n_in,
                              void* d_out, int out_size, void* d_ws, size_t ws_size,
                              hipStream_t stream) {
    const float* nf  = (const float*)d_in[0];
    const float* ef  = (const float*)d_in[1];
    const int*   src = (const int*)d_in[2];
    const int*   dst = (const int*)d_in[3];
    const float* We1 = (const float*)d_in[4];
    const float* be1 = (const float*)d_in[5];
    const float* We2 = (const float*)d_in[6];
    const float* be2 = (const float*)d_in[7];
    const float* Wa1 = (const float*)d_in[8];
    const float* ba1 = (const float*)d_in[9];
    const float* Wa2 = (const float*)d_in[10];
    const float* ba2 = (const float*)d_in[11];
    const float* Wn1 = (const float*)d_in[12];
    const float* bn1 = (const float*)d_in[13];
    const float* Wn2 = (const float*)d_in[14];
    const float* bn2 = (const float*)d_in[15];

    char* ws = (char*)d_ws;
    size_t off = 0;
    auto carve = [&](size_t bytes) {
        void* p = ws + off;
        off += (bytes + 255) & ~(size_t)255;
        return p;
    };
    unsigned short* wtWe1 = (unsigned short*)carve(160 * 128 * 2);
    unsigned short* wtWa1 = (unsigned short*)carve(160 * 128 * 2);
    unsigned short* wtWe2 = (unsigned short*)carve(128 * 64 * 2);
    unsigned short* wtWn1 = (unsigned short*)carve(128 * 128 * 2);
    unsigned short* wtWn2 = (unsigned short*)carve(128 * 64 * 2);
    float*    logits = (float*)carve((size_t)NE * 4);
    float*    exv    = (float*)carve((size_t)NE * 4);
    unsigned* mmax   = (unsigned*)carve((size_t)NN * 4);
    float*    denom  = (float*)carve((size_t)NN * 4);
    float*    agg    = (float*)carve((size_t)NN * 64 * 4);

    float* uh_n = (float*)d_out;              // [NN,64]
    float* uh_e = uh_n + (size_t)NN * 64;     // [NE,64]

    // 0) weight prep (bf16 transposed)
    mpnn_wprep<<<(160 * 128 + 255) / 256, 256, 0, stream>>>(We1, wtWe1, 160, 128);
    mpnn_wprep<<<(160 * 128 + 255) / 256, 256, 0, stream>>>(Wa1, wtWa1, 160, 128);
    mpnn_wprep<<<(128 * 64 + 255) / 256, 256, 0, stream>>>(We2, wtWe2, 128, 64);
    mpnn_wprep<<<(128 * 128 + 255) / 256, 256, 0, stream>>>(Wn1, wtWn1, 128, 128);
    mpnn_wprep<<<(128 * 64 + 255) / 256, 256, 0, stream>>>(Wn2, wtWn2, 128, 64);
    // 1) init scratch
    mpnn_init<<<(NN * 64 + 255) / 256, 256, 0, stream>>>(agg, denom, mmax);
    // 2) edge MLPs + logits + segment max
    mpnn_edge<<<NE / 64, 256, 0, stream>>>(nf, ef, src, dst,
                                           wtWe1, be1, wtWa1, ba1, wtWe2, be2,
                                           Wa2, ba2, uh_e, logits, mmax);
    // 3) softmax numerator + denom
    mpnn_softmax<<<(NE + 255) / 256, 256, 0, stream>>>(logits, dst, mmax, exv, denom);
    // 4) attention-weighted scatter-sum
    mpnn_agg<<<(NE * 64) / 256, 256, 0, stream>>>(exv, denom, dst, uh_e, agg);
    // 5) node MLP
    mpnn_node<<<(NN + 63) / 64, 256, 0, stream>>>(agg, nf, wtWn1, bn1, wtWn2, bn2, uh_n);
}
